// Intention_PS_LSTM_84988812853644
// MI455X (gfx1250) — compile-verified, matrix operand order checked by reading, not run on hardware
//
#include <hip/hip_runtime.h>
#include <hip/hip_bf16.h>

typedef __bf16 v16bf __attribute__((ext_vector_type(16)));
typedef float v8f __attribute__((ext_vector_type(8)));

#define TSTEPS 16
#define HID 512
#define GATES 2048
#define HBSTR 520          // bf16 h row stride (elements); 520*2 bytes = 16B-aligned rows

// ---------------- helpers ----------------

__device__ __forceinline__ unsigned short f2bf(float f) {
  union { float f; unsigned u; } a; a.f = f;
  unsigned u = a.u;
  unsigned r = u + 0x7FFFu + ((u >> 16) & 1u);   // round-to-nearest-even
  return (unsigned short)(r >> 16);
}
__device__ __forceinline__ float bf2f(unsigned short s) {
  union { unsigned u; float f; } a; a.u = ((unsigned)s) << 16;
  return a.f;
}
__device__ __forceinline__ float sigm(float x) {
  return __builtin_amdgcn_rcpf(1.0f + __expf(-x));
}
__device__ __forceinline__ float tanh_f(float x) {
  float e = __expf(-2.0f * x);
  return (1.0f - e) * __builtin_amdgcn_rcpf(1.0f + e);
}
__device__ __forceinline__ int kbase(int v, int half) {
  return ((v & 4) ? 16 : 0) + half * 8 + (v & 3) * 2;
}

// ---------------- weight pre-pack: W_hh [2048,512] f32 -> bf16 B-fragments ----
// frag id = nt*16 + kt; within a fragment lane l owns 16 contiguous bf16
// (its v16bf register image). B[k][n] = W[n][k].

extern "C" __global__ __launch_bounds__(256)
void pack_whh_kernel(const float* __restrict__ W0,
                     const float* __restrict__ W1,
                     const float* __restrict__ W2,
                     unsigned short* __restrict__ out) {
  int idx = blockIdx.x * blockDim.x + threadIdx.x;   // one (frag,lane) per thread
  int mat = idx >> 16;                               // 65536 = 128*16*32 per matrix
  int fi  = idx & 65535;
  const float* W = (mat == 0) ? W0 : ((mat == 1) ? W1 : W2);
  unsigned short* o = out + (size_t)mat * (GATES * HID) + (size_t)fi * 16;
  int nt   = fi >> 9;
  int kt   = (fi >> 5) & 15;
  int lane = fi & 31;
  int n    = nt * 16 + (lane & 15);
  int half = lane >> 4;
#pragma unroll
  for (int e = 0; e < 16; ++e) {
    int v = e >> 1, p = e & 1;
    int k = kt * 32 + kbase(v, half) + p;
    o[e] = f2bf(W[n * HID + k]);
  }
}

// ---------------- one LSTM step for a 16-row batch tile ----------------
// h_bf: bf16 [16][HBSTR] (read as A via 2x ds_load_b128, then overwritten)
// c[4]: per-wave cell state tiles in WMMA D layout
// wave w owns unit tiles u = w*4 .. w*4+3; gate tile N = g*512 + u*16

__device__ __forceinline__ void lstm_step(
    const unsigned short* __restrict__ pwl,  // PW + l*16 (per-lane B base)
    const float* __restrict__ x_t,           // LDS [16][4]
    unsigned short* __restrict__ h_bf,
    const float* __restrict__ wih_lds,       // [2048][4]
    const float* __restrict__ bias_lds,      // [2048]
    v8f c[4], int w, int l) {
  const int half = l >> 4;
  const int mcol = l & 15;

  // stage this lane's x rows (D-layout rows: m = r + 8*half)
  float4 xr[8];
#pragma unroll
  for (int r = 0; r < 8; ++r)
    xr[r] = *(const float4*)&x_t[(r + half * 8) * 4];

  // acc init: bias + x @ W_ih.T  (K = 4, VALU)
  v8f acc[4][4];
#pragma unroll
  for (int ul = 0; ul < 4; ++ul) {
    int u = w * 4 + ul;
#pragma unroll
    for (int g = 0; g < 4; ++g) {
      int n = g * 512 + u * 16 + mcol;
      float bb = bias_lds[n];
      float4 wv = *(const float4*)&wih_lds[n * 4];
      v8f a;
#pragma unroll
      for (int r = 0; r < 8; ++r)
        a[r] = bb + xr[r].x * wv.x + xr[r].y * wv.y + xr[r].z * wv.z + xr[r].w * wv.w;
      acc[ul][g] = a;
    }
  }

  // K loop: A = 2x ds_load_b128 of bf16 h; B = immediate-offset global loads.
  // Unroll 2 so consecutive iterations use distinct A-fragment registers,
  // removing the WMMA->ds_load WAR hazard NOPs at the loop head.
  const unsigned short* arow = &h_bf[mcol * HBSTR + half * 8];
#pragma unroll 2
  for (int kt = 0; kt < 16; ++kt) {
    union { uint4 q[2]; v16bf b; } afr;
    const unsigned short* ar = arow + kt * 32;
    afr.q[0] = *(const uint4*)ar;          // frag elems 0..7  (k = +0..7)
    afr.q[1] = *(const uint4*)(ar + 16);   // frag elems 8..15 (k = +16..23)

    const unsigned short* pk = pwl + kt * 512;   // fragment stride = 512 elems
#pragma unroll
    for (int ul = 0; ul < 4; ++ul) {
      int u = w * 4 + ul;
#pragma unroll
      for (int g = 0; g < 4; ++g) {
        int nt = g * 32 + u;
        v16bf bfr = *(const v16bf*)(pk + nt * 8192);   // const byte offsets
        acc[ul][g] = __builtin_amdgcn_wmma_f32_16x16x32_bf16(
            false, afr.b, false, bfr, (short)0, acc[ul][g], false, false);
      }
    }
  }

  __syncthreads();   // all A reads of h_bf done before overwrite

  // cell update + write new h as bf16 (D layout: row m = r + 8*half)
#pragma unroll
  for (int ul = 0; ul < 4; ++ul) {
    int ku = (w * 4 + ul) * 16 + mcol;
#pragma unroll
    for (int r = 0; r < 8; ++r) {
      float gi = sigm(acc[ul][0][r]);
      float gf = sigm(acc[ul][1][r]);
      float gg = tanh_f(acc[ul][2][r]);
      float go = sigm(acc[ul][3][r]);
      float cc = gf * c[ul][r] + gi * gg;
      c[ul][r] = cc;
      h_bf[(r + 8 * half) * HBSTR + ku] = f2bf(go * tanh_f(cc));
    }
  }
  __syncthreads();
}

// ---------------- fused: speed-LSTM + pos-LSTM + decoder ----------------

extern "C" __global__ __launch_bounds__(256)
void lstm_fused_kernel(
    const float* __restrict__ speed, const float* __restrict__ pos,
    const float* __restrict__ Ws_ih, const float* __restrict__ bs_ih, const float* __restrict__ bs_hh,
    const float* __restrict__ Wp_ih, const float* __restrict__ bp_ih, const float* __restrict__ bp_hh,
    const float* __restrict__ Wd_ih, const float* __restrict__ bd_ih, const float* __restrict__ bd_hh,
    const float* __restrict__ W_fc,  const float* __restrict__ b_fc,
    const float* __restrict__ W_emb, const float* __restrict__ b_emb,
    const unsigned short* __restrict__ Ps,
    const unsigned short* __restrict__ Pp,
    const unsigned short* __restrict__ Pd,
    float* __restrict__ out) {
  __shared__ unsigned short h_bf[16 * HBSTR];
  __shared__ unsigned short hs_bf[16 * HBSTR];
  __shared__ float cs_lds[16 * HID];
  __shared__ float wih_lds[GATES * 4];
  __shared__ float bias_lds[GATES];
  __shared__ float x_lds[TSTEPS * 16 * 4];   // [t][m][4]; slot 0 reused by decoder
  __shared__ float wfc_lds[2 * HID];
  __shared__ float cross_lds[16 * 2];

  const int tid  = threadIdx.x;
  const int w    = tid >> 5;
  const int l    = tid & 31;
  const int half = l >> 4;
  const int mcol = l & 15;
  const int wgRow = blockIdx.x * 16;

  const unsigned short* pwl_s = Ps + l * 16;   // per-lane B-fragment bases
  const unsigned short* pwl_p = Pp + l * 16;
  const unsigned short* pwl_d = Pd + l * 16;

  v8f c[4];

  // === speed encoder ===
  for (int i = tid; i < GATES; i += 256) {
    bias_lds[i] = bs_ih[i] + bs_hh[i];
    *(float4*)&wih_lds[i * 4] = *(const float4*)&Ws_ih[i * 4];
  }
  {
    int t = tid >> 4, m = tid & 15;
    *(float4*)&x_lds[tid * 4] =
        *(const float4*)&speed[((size_t)(wgRow + m) * TSTEPS + t) * 4];
  }
  for (int i = tid; i < 16 * HBSTR; i += 256) h_bf[i] = 0;
#pragma unroll
  for (int ul = 0; ul < 4; ++ul)
#pragma unroll
    for (int r = 0; r < 8; ++r) c[ul][r] = 0.0f;
  __syncthreads();

  for (int t = 0; t < TSTEPS; ++t)
    lstm_step(pwl_s, &x_lds[t * 64], h_bf, wih_lds, bias_lds, c, w, l);

  // save speed state
  for (int i = tid; i < 16 * HBSTR; i += 256) hs_bf[i] = h_bf[i];
#pragma unroll
  for (int ul = 0; ul < 4; ++ul) {
    int ku = (w * 4 + ul) * 16 + mcol;
#pragma unroll
    for (int r = 0; r < 8; ++r) cs_lds[(r + 8 * half) * HID + ku] = c[ul][r];
  }
  __syncthreads();

  // === pos encoder ===
  for (int i = tid; i < GATES; i += 256) {
    bias_lds[i] = bp_ih[i] + bp_hh[i];
    *(float4*)&wih_lds[i * 4] = *(const float4*)&Wp_ih[i * 4];
  }
  {
    int t = tid >> 4, m = tid & 15;
    *(float4*)&x_lds[tid * 4] =
        *(const float4*)&pos[((size_t)(wgRow + m) * TSTEPS + t) * 4];
  }
  for (int i = tid; i < 16 * HBSTR; i += 256) h_bf[i] = 0;
#pragma unroll
  for (int ul = 0; ul < 4; ++ul)
#pragma unroll
    for (int r = 0; r < 8; ++r) c[ul][r] = 0.0f;
  __syncthreads();

  for (int t = 0; t < TSTEPS; ++t)
    lstm_step(pwl_p, &x_lds[t * 64], h_bf, wih_lds, bias_lds, c, w, l);

  // === fuse states: h0 = hp + hs ; c0 = cp + cs ; last_pos = pos[:, -1, :] ===
  for (int i = tid; i < 16 * HBSTR; i += 256)
    h_bf[i] = f2bf(bf2f(h_bf[i]) + bf2f(hs_bf[i]));
#pragma unroll
  for (int ul = 0; ul < 4; ++ul) {
    int ku = (w * 4 + ul) * 16 + mcol;
#pragma unroll
    for (int r = 0; r < 8; ++r) c[ul][r] += cs_lds[(r + 8 * half) * HID + ku];
  }
  if (tid < 64) x_lds[tid] = x_lds[15 * 64 + tid];   // last_pos -> decoder x slot
  for (int i = tid; i < GATES; i += 256) {
    bias_lds[i] = bd_ih[i] + bd_hh[i];
    *(float4*)&wih_lds[i * 4] = *(const float4*)&Wd_ih[i * 4];
  }
  for (int i = tid; i < 2 * HID; i += 256) wfc_lds[i] = W_fc[i];
  __syncthreads();

  // === decoder ===
  for (int s = 0; s < TSTEPS; ++s) {
    lstm_step(pwl_d, x_lds, h_bf, wih_lds, bias_lds, c, w, l);

    // crossing = relu(h @ W_fc.T + b_fc)   [16 rows x 2]
    if (tid < 32) {
      int m = tid >> 1, j = tid & 1;
      float acc = b_fc[j];
      const unsigned short* hr = &h_bf[m * HBSTR];
      const float* wr = &wfc_lds[j * HID];
#pragma unroll 4
      for (int k = 0; k < HID; ++k) acc += bf2f(hr[k]) * wr[k];
      cross_lds[m * 2 + j] = fmaxf(acc, 0.0f);
    }
    __syncthreads();

    // softmax -> out ; emb -> next last_pos
    if (tid < 16) {
      int m = tid;
      float c0 = cross_lds[m * 2], c1 = cross_lds[m * 2 + 1];
      float mx = fmaxf(c0, c1);
      float e0 = __expf(c0 - mx), e1 = __expf(c1 - mx);
      float inv = __builtin_amdgcn_rcpf(e0 + e1);
      size_t b = wgRow + m;
      float2 pr; pr.x = e0 * inv; pr.y = e1 * inv;
      *(float2*)&out[(b * TSTEPS + s) * 2] = pr;
#pragma unroll
      for (int i = 0; i < 4; ++i) {
        float v = b_emb[i] + c0 * W_emb[i * 2] + c1 * W_emb[i * 2 + 1];
        x_lds[m * 4 + i] = fmaxf(v, 0.0f);
      }
    }
    __syncthreads();
  }
}

// ---------------- host launcher ----------------

extern "C" void kernel_launch(void* const* d_in, const int* in_sizes, int n_in,
                              void* d_out, int out_size, void* d_ws, size_t ws_size,
                              hipStream_t stream) {
  const float* speed = (const float*)d_in[0];
  const float* pos   = (const float*)d_in[1];
  const float* Ws_ih = (const float*)d_in[2];
  const float* Ws_hh = (const float*)d_in[3];
  const float* bs_ih = (const float*)d_in[4];
  const float* bs_hh = (const float*)d_in[5];
  const float* Wp_ih = (const float*)d_in[6];
  const float* Wp_hh = (const float*)d_in[7];
  const float* bp_ih = (const float*)d_in[8];
  const float* bp_hh = (const float*)d_in[9];
  const float* Wd_ih = (const float*)d_in[10];
  const float* Wd_hh = (const float*)d_in[11];
  const float* bd_ih = (const float*)d_in[12];
  const float* bd_hh = (const float*)d_in[13];
  const float* W_fc  = (const float*)d_in[14];
  const float* b_fc  = (const float*)d_in[15];
  const float* W_emb = (const float*)d_in[16];
  const float* b_emb = (const float*)d_in[17];

  const int B = in_sizes[0] / (TSTEPS * 4);   // 8192

  unsigned short* packed = (unsigned short*)d_ws;   // 3 x 2048*512 bf16 = 6 MB
  unsigned short* Ps = packed + 0 * (size_t)(GATES * HID);
  unsigned short* Pp = packed + 1 * (size_t)(GATES * HID);
  unsigned short* Pd = packed + 2 * (size_t)(GATES * HID);

  // pack all three W_hh matrices into WMMA bf16 B-fragment layout
  {
    int total = 3 * 128 * 16 * 32;   // (mat, nt, kt, lane)
    pack_whh_kernel<<<total / 256, 256, 0, stream>>>(Ws_hh, Wp_hh, Wd_hh, packed);
  }

  // fused persistent pipeline: one workgroup per 16 batch rows
  lstm_fused_kernel<<<B / 16, 256, 0, stream>>>(
      speed, pos,
      Ws_ih, bs_ih, bs_hh,
      Wp_ih, bp_ih, bp_hh,
      Wd_ih, bd_ih, bd_hh,
      W_fc, b_fc, W_emb, b_emb,
      Ps, Pp, Pd,
      (float*)d_out);
}
